// GModule_81939386073329
// MI455X (gfx1250) — matrix operand order repaired and checked by reading, not compiled, not run on hardware
//
#include <hip/hip_runtime.h>
#include <hip/hip_bf16.h>

// ---------------------------------------------------------------------------
// MI455X (gfx1250) implementation of the SIGMA-style graph-matching head.
// All GEMMs: v_wmma_f32_16x16x32_bf16, double-buffered LDS staging via
// GLOBAL_LOAD_ASYNC_TO_LDS_B128 (ASYNCcnt) when the toolchain exposes it.
// Every GEMM is C = A * B^T with B stored K-major (weights pre-transposed
// once; V transposed per attention) so all LDS fills are plain 16B copies.
// ---------------------------------------------------------------------------

typedef __attribute__((ext_vector_type(16))) __bf16 v16bf;
typedef __attribute__((ext_vector_type(8)))  float  v8f;
typedef __attribute__((__vector_size__(16))) int    vi4;   // int4 vector (16B)

using u16 = unsigned short;

#define AS1 __attribute__((address_space(1)))
#define AS3 __attribute__((address_space(3)))

#if defined(__gfx1250__) && \
    __has_builtin(__builtin_amdgcn_global_load_async_to_lds_b128) && \
    __has_builtin(__builtin_amdgcn_s_wait_asynccnt)
#define HAS_ASYNC 1
#else
#define HAS_ASYNC 0
#endif

__device__ __forceinline__ u16 f32_to_bf16(float f) {
  union { float f; unsigned int u; } a; a.f = f;
  unsigned int u = a.u;
  u += 0x7FFFu + ((u >> 16) & 1u);   // round-to-nearest-even
  return (u16)(u >> 16);
}

// ---------------------------------------------------------------------------
// Conversion / transpose kernels
// ---------------------------------------------------------------------------
__global__ __launch_bounds__(256) void k_cvt_bf16(const float* __restrict__ x,
                                                  u16* __restrict__ y,
                                                  long long n) {
  long long i = (long long)blockIdx.x * 256 + threadIdx.x;
  long long stride = (long long)gridDim.x * 256;
  for (; i < n; i += stride) y[i] = f32_to_bf16(x[i]);
}

// y[C,R] = bf16(x[R,C]^T)   (coalesced writes)
__global__ __launch_bounds__(256) void k_cvt_t_bf16(const float* __restrict__ x,
                                                    u16* __restrict__ y,
                                                    int R, int C) {
  long long o = (long long)blockIdx.x * 256 + threadIdx.x;
  long long n = (long long)R * C;
  if (o >= n) return;
  int c = (int)(o / R), r = (int)(o % R);
  y[o] = f32_to_bf16(x[(long long)r * C + c]);
}

// y[C,R] = x[R,C]^T for bf16 payloads (coalesced writes)
__global__ __launch_bounds__(256) void k_t16(const u16* __restrict__ x,
                                             u16* __restrict__ y, int R, int C) {
  long long o = (long long)blockIdx.x * 256 + threadIdx.x;
  long long n = (long long)R * C;
  if (o >= n) return;
  int c = (int)(o / R), r = (int)(o % R);
  y[o] = x[(long long)r * C + c];
}

// A_sym = 0.5*(A + A^T) -> bf16 (symmetric, so it is its own K-major form)
__global__ __launch_bounds__(256) void k_sym_bf16(const float* __restrict__ A,
                                                  u16* __restrict__ y, int D) {
  long long idx = (long long)blockIdx.x * 256 + threadIdx.x;
  long long n = (long long)D * D;
  if (idx >= n) return;
  int i = (int)(idx / D), j = (int)(idx % D);
  y[idx] = f32_to_bf16(0.5f * (A[(long long)i * D + j] + A[(long long)j * D + i]));
}

// ---------------------------------------------------------------------------
// WMMA GEMM: C[M,N] = scale * (A @ BT^T) + bias
//   A : [M,K] bf16 row-major
//   BT: [N,K] bf16 row-major (i.e. logical B^T, K contiguous)
//   C : f32 or bf16, row-major [M,N]
// Block: 256 threads = 8 waves; tile 128x128; wave tile 32x64 (2x4 wmma).
// Double-buffered LDS, async global->LDS staging.
// ---------------------------------------------------------------------------
#define TILE_M 128
#define TILE_N 128
#define TILE_K 32
#define LDS_P  40   // LDS row pitch (elements): 80B rows, 16B-aligned chunks

template <bool OUT_F32>
__global__ __launch_bounds__(256) void k_gemm(
    const u16* __restrict__ A, const u16* __restrict__ BT,
    void* __restrict__ C, const float* __restrict__ bias,
    float scale, int M, int N, int K) {
  // lds[buf][0] = A tile [128][40], lds[buf][1] = BT tile [128][40]
  __shared__ alignas(16) u16 lds[2][2][TILE_M * LDS_P];

  const int tid  = threadIdx.x;
  const int lane = tid & 31;
  const int wid  = tid >> 5;
  const int wm   = (wid & 3) * 32;   // wave row base within tile (0..96)
  const int wn   = (wid >> 2) * 64;  // wave col base within tile (0 or 64)
  const int bm   = blockIdx.y * TILE_M;
  const int bn   = blockIdx.x * TILE_N;
  const int rr   = lane & 15;
  const int hh   = lane >> 4;

  v8f acc[2][4] = {};

  // stage K-tile `kb` into buffer `buf`: 4 async 16B loads per thread
  auto stage = [&](int kb, int buf) {
    u16* la = &lds[buf][0][0];
    u16* lb = &lds[buf][1][0];
#pragma unroll
    for (int i = 0; i < 2; ++i) {
      int idx = tid + i * 256;
      int r   = idx >> 2;          // 0..127
      int c8  = (idx & 3) * 8;     // 0,8,16,24
      const u16* ga = A  + (long long)(bm + r) * K + kb + c8;
      const u16* gb = BT + (long long)(bn + r) * K + kb + c8;
      u16* pa = la + r * LDS_P + c8;
      u16* pb = lb + r * LDS_P + c8;
#if HAS_ASYNC
      __builtin_amdgcn_global_load_async_to_lds_b128((AS1 vi4*)ga, (AS3 vi4*)pa, 0, 0);
      __builtin_amdgcn_global_load_async_to_lds_b128((AS1 vi4*)gb, (AS3 vi4*)pb, 0, 0);
#else
      *(uint4*)pa = *(const uint4*)ga;
      *(uint4*)pb = *(const uint4*)gb;
#endif
    }
  };

  union F16 { uint4 u[2]; v16bf v; };

  auto compute = [&](int buf) {
    const u16* la = &lds[buf][0][0];
    const u16* lb = &lds[buf][1][0];
    v16bf af[2], bf[4];
#pragma unroll
    for (int mi = 0; mi < 2; ++mi) {
      // A 16x32 frag: lane<16 -> row rr, K 0..7 & 16..23; lane>=16 -> K 8..15 & 24..31
      F16 t;
      t.u[0] = *(const uint4*)(la + (wm + mi * 16 + rr) * LDS_P + hh * 8);
      t.u[1] = *(const uint4*)(la + (wm + mi * 16 + rr) * LDS_P + 16 + hh * 8);
      af[mi] = t.v;
    }
#pragma unroll
    for (int ni = 0; ni < 4; ++ni) {
      // B 32x16 frag: col = lane%16; lane<16 -> K 0..7 & 16..23; lane>=16 -> K 8..15 & 24..31
      F16 t;
      t.u[0] = *(const uint4*)(lb + (wn + ni * 16 + rr) * LDS_P + hh * 8);
      t.u[1] = *(const uint4*)(lb + (wn + ni * 16 + rr) * LDS_P + 16 + hh * 8);
      bf[ni] = t.v;
    }
#pragma unroll
    for (int mi = 0; mi < 2; ++mi)
#pragma unroll
      for (int ni = 0; ni < 4; ++ni)
        acc[mi][ni] = __builtin_amdgcn_wmma_f32_16x16x32_bf16(
            false, af[mi], false, bf[ni], (short)0, acc[mi][ni], false, false);
  };

  const int niter = K / TILE_K;
  stage(0, 0);
  for (int it = 0; it < niter; ++it) {
    if (it + 1 < niter) {
      stage((it + 1) * TILE_K, (it + 1) & 1);
#if HAS_ASYNC
      __builtin_amdgcn_s_wait_asynccnt(4);  // tile `it` landed; next in flight
#endif
    } else {
#if HAS_ASYNC
      __builtin_amdgcn_s_wait_asynccnt(0);
#endif
    }
    __syncthreads();        // tile `it` visible to all waves
    compute(it & 1);
    __syncthreads();        // all waves done reading before buffer reuse
  }

  // ---- epilogue: C/D layout: VGPR e -> row e + 8*(lane/16), col lane%16 ----
#pragma unroll
  for (int mi = 0; mi < 2; ++mi)
#pragma unroll
    for (int ni = 0; ni < 4; ++ni) {
      const int col = bn + wn + ni * 16 + rr;
      const float bv = bias ? bias[col] : 0.0f;
#pragma unroll
      for (int e = 0; e < 8; ++e) {
        const int row = bm + wm + mi * 16 + hh * 8 + e;
        const float v = acc[mi][ni][e] * scale + bv;
        if constexpr (OUT_F32)
          ((float*)C)[(long long)row * N + col] = v;
        else
          ((u16*)C)[(long long)row * N + col] = f32_to_bf16(v);
      }
    }
}

// ---------------------------------------------------------------------------
// Row softmax: S [rows, N] f32 -> P [rows, N] bf16. One block per row.
// ---------------------------------------------------------------------------
__global__ __launch_bounds__(256) void k_softmax(const float* __restrict__ S,
                                                 u16* __restrict__ P, int N) {
  __shared__ float red[256];
  const int t = threadIdx.x;
  const float* s = S + (long long)blockIdx.x * N;
  u16* p = P + (long long)blockIdx.x * N;

  float mx = -3.4e38f;
  for (int c = t; c < N; c += 256) mx = fmaxf(mx, s[c]);
  red[t] = mx;
  __syncthreads();
  for (int o = 128; o > 0; o >>= 1) {
    if (t < o) red[t] = fmaxf(red[t], red[t + o]);
    __syncthreads();
  }
  mx = red[0];
  __syncthreads();

  float sum = 0.f;
  for (int c = t; c < N; c += 256) sum += __expf(s[c] - mx);
  red[t] = sum;
  __syncthreads();
  for (int o = 128; o > 0; o >>= 1) {
    if (t < o) red[t] += red[t + o];
    __syncthreads();
  }
  const float inv = 1.0f / red[0];
  for (int c = t; c < N; c += 256) p[c] = f32_to_bf16(__expf(s[c] - mx) * inv);
}

// ---------------------------------------------------------------------------
// Instance-norm reductions (deterministic tree, double accumulators)
// ---------------------------------------------------------------------------
__global__ __launch_bounds__(256) void k_red_partial(const float* __restrict__ X,
                                                     double* __restrict__ part,
                                                     long long n) {
  __shared__ double rs[256], rq[256];
  const int t = threadIdx.x;
  double s = 0.0, q = 0.0;
  for (long long i = (long long)blockIdx.x * 256 + t; i < n;
       i += (long long)gridDim.x * 256) {
    double v = (double)X[i];
    s += v;
    q += v * v;
  }
  rs[t] = s; rq[t] = q;
  __syncthreads();
  for (int o = 128; o > 0; o >>= 1) {
    if (t < o) { rs[t] += rs[t + o]; rq[t] += rq[t + o]; }
    __syncthreads();
  }
  if (t == 0) { part[2 * blockIdx.x] = rs[0]; part[2 * blockIdx.x + 1] = rq[0]; }
}

__global__ __launch_bounds__(256) void k_red_final(const double* __restrict__ part,
                                                   int nparts,
                                                   float* __restrict__ stats,
                                                   double n, float eps) {
  __shared__ double rs[256], rq[256];
  const int t = threadIdx.x;
  double s = 0.0, q = 0.0;
  for (int i = t; i < nparts; i += 256) { s += part[2 * i]; q += part[2 * i + 1]; }
  rs[t] = s; rq[t] = q;
  __syncthreads();
  for (int o = 128; o > 0; o >>= 1) {
    if (t < o) { rs[t] += rs[t + o]; rq[t] += rq[t + o]; }
    __syncthreads();
  }
  if (t == 0) {
    double mu  = rs[0] / n;
    double var = rq[0] / n - mu * mu;
    stats[0] = (float)mu;
    stats[1] = (float)(1.0 / sqrt(var + (double)eps));
  }
}

__global__ __launch_bounds__(256) void k_instnorm(float* __restrict__ X,
                                                  const float* __restrict__ stats,
                                                  long long n) {
  const float mu = stats[0], rsig = stats[1];
  long long i = (long long)blockIdx.x * 256 + threadIdx.x;
  long long stride = (long long)gridDim.x * 256;
  for (; i < n; i += stride) X[i] = (X[i] - mu) * rsig;
}

// ---------------------------------------------------------------------------
// Host orchestration
// ---------------------------------------------------------------------------
extern "C" void kernel_launch(void* const* d_in, const int* in_sizes, int n_in,
                              void* d_out, int out_size, void* d_ws, size_t ws_size,
                              hipStream_t stream) {
  (void)in_sizes; (void)n_in; (void)out_size; (void)ws_size;
  const int NN = 4096, D = 1024, KIN = 2048;
  const float SCALE = 0.03125f;  // 1/sqrt(1024)

  const float* nodes_1 = (const float*)d_in[0];
  const float* nodes_2 = (const float*)d_in[1];
  const float* W_in    = (const float*)d_in[2];
  const float* b_in    = (const float*)d_in[3];
  const float* Wf[8]   = { (const float*)d_in[4],  (const float*)d_in[5],
                           (const float*)d_in[6],  (const float*)d_in[7],
                           (const float*)d_in[8],  (const float*)d_in[9],
                           (const float*)d_in[10], (const float*)d_in[11] };
  const float* A_aff   = (const float*)d_in[12];
  float* out = (float*)d_out;

  // --- workspace carve-up ---
  char* w = (char*)d_ws;
  size_t off = 0;
  auto take = [&](size_t bytes) -> char* {
    char* p = w + off;
    off += (bytes + 255) & ~(size_t)255;
    return p;
  };
  double* partials = (double*)take(2048 * sizeof(double));
  float*  stats    = (float*)take(256);
  u16* wWinT = (u16*)take((size_t)KIN * D * 2);   // W_in^T [D,KIN]
  u16* wbT[8];                                    // W^T [D,D]
  for (int i = 0; i < 8; ++i) wbT[i] = (u16*)take((size_t)D * D * 2);
  u16* wAs  = (u16*)take((size_t)D * D * 2);      // A_sym (symmetric)
  u16* n1   = (u16*)take((size_t)NN * D * 2);
  u16* n2   = (u16*)take((size_t)NN * D * 2);
  u16* Qb   = (u16*)take((size_t)NN * D * 2);
  u16* Kb   = (u16*)take((size_t)NN * D * 2);
  u16* Vb   = (u16*)take((size_t)NN * D * 2);
  u16* VT   = (u16*)take((size_t)NN * D * 2);     // V^T [D,NN]
  u16* t0   = (u16*)take((size_t)NN * D * 2);
  u16* n1c  = (u16*)take((size_t)NN * D * 2);
  u16* n2c  = (u16*)take((size_t)NN * D * 2);
  float* S  = (float*)take((size_t)NN * NN * 4);
  u16* P    = (u16*)take((size_t)NN * NN * 2);
  // bf16 node copies live in S's region (dead once S is first needed)
  u16* nA = (u16*)S;
  u16* nB = (u16*)S + (size_t)NN * KIN;

  // C[M,N] = scale*(A[M,K] @ BT[N,K]^T) + bias
  auto gemm = [&](const u16* A, const u16* BT, void* C, const float* bias,
                  float scale, int M, int Nc, int K, bool outF32) {
    dim3 g(Nc / TILE_N, M / TILE_M), b(256);
    if (outF32) k_gemm<true ><<<g, b, 0, stream>>>(A, BT, C, bias, scale, M, Nc, K);
    else        k_gemm<false><<<g, b, 0, stream>>>(A, BT, C, bias, scale, M, Nc, K);
  };

  auto cvt = [&](const float* x, u16* y, long long n) {
    k_cvt_bf16<<<(unsigned)((n + 255) / 256), 256, 0, stream>>>(x, y, n);
  };
  auto cvt_t = [&](const float* x, u16* y, int R, int C) {
    long long n = (long long)R * C;
    k_cvt_t_bf16<<<(unsigned)((n + 255) / 256), 256, 0, stream>>>(x, y, R, C);
  };

  // ---- stage 0: convert weights (transposed to K-major), A_sym, nodes ----
  cvt_t(W_in, wWinT, KIN, D);
  for (int i = 0; i < 8; ++i) cvt_t(Wf[i], wbT[i], D, D);
  k_sym_bf16<<<(unsigned)(((long long)D * D + 255) / 256), 256, 0, stream>>>(A_aff, wAs, D);
  cvt(nodes_1, nA, (long long)NN * KIN);
  cvt(nodes_2, nB, (long long)NN * KIN);

  // ---- stage 1: input projection n = nodes @ W_in + b_in ----
  gemm(nA, wWinT, n1, b_in, 1.0f, NN, D, KIN, false);
  gemm(nB, wWinT, n2, b_in, 1.0f, NN, D, KIN, false);

  // ---- attention: res = softmax(qn Wq (kvn Wk)^T * scale) (kvn Wv) Wo ----
  auto attn = [&](const u16* qn, const u16* kvn, const u16* wqT, const u16* wkT,
                  const u16* wvT, const u16* woT, u16* res) {
    gemm(qn,  wqT, Qb, nullptr, 1.0f, NN, D, D, false);
    gemm(kvn, wkT, Kb, nullptr, 1.0f, NN, D, D, false);
    gemm(kvn, wvT, Vb, nullptr, 1.0f, NN, D, D, false);
    k_t16<<<(unsigned)(((long long)NN * D + 255) / 256), 256, 0, stream>>>(Vb, VT, NN, D);
    gemm(Qb, Kb, S, nullptr, SCALE, NN, NN, D, true);   // S = Q K^T * scale
    k_softmax<<<NN, 256, 0, stream>>>(S, P, NN);
    gemm(P,  VT,  t0,  nullptr, 1.0f, NN, D, NN, false); // P @ V
    gemm(t0, woT, res, nullptr, 1.0f, NN, D, D,  false); // @ Wo
  };

  // ---- stage 2: intra-domain self attention (shared weights) ----
  attn(n1, n1, wbT[0], wbT[1], wbT[2], wbT[3], n1);
  attn(n2, n2, wbT[0], wbT[1], wbT[2], wbT[3], n2);

  // ---- stage 3: cross-domain attention (shared weights) ----
  attn(n1, n2, wbT[4], wbT[5], wbT[6], wbT[7], n1c);
  attn(n2, n1, wbT[4], wbT[5], wbT[6], wbT[7], n2c);

  // ---- stage 4: affinity M = (n1_c @ A_sym) @ n2_c^T ----
  gemm(n1c, wAs, t0, nullptr, 1.0f, NN, D, D, false);   // A_sym^T == A_sym
  gemm(t0, n2c, out, nullptr, 1.0f, NN, NN, D, true);

  // ---- stage 5: InstanceNorm2d over whole matrix ----
  const long long total = (long long)NN * NN;
  k_red_partial<<<1024, 256, 0, stream>>>(out, partials, total);
  k_red_final<<<1, 256, 0, stream>>>(partials, 1024, stats, (double)total, 1e-5f);
  k_instnorm<<<2048, 256, 0, stream>>>(out, stats, total);
}
